// CenterClusterLoss_47201690583526
// MI455X (gfx1250) — compile-verified, matrix operand order checked by reading, not run on hardware
//
#include <hip/hip_runtime.h>
#include <hip/hip_bf16.h>

#define B_N 1024
#define P_N 8
#define D_N 2048
#define PD  (P_N * D_N)      // 16384
#define NUM_IDS 64
#define MARGIN_F 0.3f
#define EPS_F 1e-12f

#define TM 64                // block tile M (rows I)
#define TN 128               // block tile N (rows J)
#define TK 64                // K tile
#define NKT (D_N / TK)       // 32 k-tiles per p
#define LDS_STRIDE 72        // padded bf16 row stride (conflict-free b128 reads)
#define A_BUF_BYTES (TM * LDS_STRIDE * 2)   // 9216
#define B_BUF_BYTES (TN * LDS_STRIDE * 2)   // 18432

typedef __attribute__((ext_vector_type(16))) __bf16 v16bf;
typedef __attribute__((ext_vector_type(8)))  float  v8f;

__device__ __forceinline__ unsigned short f2bf(float x) {
    unsigned int u = __float_as_uint(x);
    u += 0x7FFFu + ((u >> 16) & 1u);          // round-to-nearest-even
    return (unsigned short)(u >> 16);
}
__device__ __forceinline__ float bf2f(unsigned short b) {
    return __uint_as_float(((unsigned int)b) << 16);
}

// 32-bit LDS byte offset of a shared-memory pointer (generic -> AS3 -> int)
__device__ __forceinline__ unsigned lds_off_of(const void* p) {
    return (unsigned)(size_t)(const __attribute__((address_space(3))) void*)p;
}
// CDNA5 async copy, GVS form: mem = SGPR64 base + per-lane i32 byte offset.
// 16 bytes per lane into LDS, tracked by ASYNCcnt.
__device__ __forceinline__ void async_b128_s(unsigned lds_addr, unsigned voff,
                                             unsigned long long sbase) {
    asm volatile("global_load_async_to_lds_b128 %0, %1, %2"
                 :: "v"(lds_addr), "v"(voff), "s"(sbase)
                 : "memory");
}
__device__ __forceinline__ void wait_async0() {
    asm volatile("s_wait_asynccnt 0" ::: "memory");
}

// ---------------- kernel 0: f32 feats -> bf16 copy (memory bound, once) ----
__global__ void k_tobf16(const float* __restrict__ feats, unsigned short* __restrict__ fb) {
    size_t idx = ((size_t)blockIdx.x * 256 + threadIdx.x) * 4;
    float4 v = *(const float4*)&feats[idx];
    unsigned int lo = (unsigned int)f2bf(v.x) | ((unsigned int)f2bf(v.y) << 16);
    unsigned int hi = (unsigned int)f2bf(v.z) | ((unsigned int)f2bf(v.w) << 16);
    *(uint2*)&fb[idx] = make_uint2(lo, hi);
}

// ---------------- kernel 1: per-class counts (deterministic) ----------------
__global__ void k_counts(const int* __restrict__ labels, float* __restrict__ countsf) {
    int c = threadIdx.x;                       // 64 threads
    int n = 0;
    for (int b = 0; b < B_N; ++b) n += (labels[b] == c) ? 1 : 0;
    countsf[c] = (float)n;
}

// ---------------- kernel 2: segment-mean centers (f32, deterministic) -------
__global__ void k_centers(const float* __restrict__ feats, const int* __restrict__ labels,
                          const float* __restrict__ countsf, float* __restrict__ centers) {
    __shared__ int slab[B_N];
    int t = threadIdx.x;
    for (int b = t; b < B_N; b += 256) slab[b] = labels[b];
    __syncthreads();
    int id  = blockIdx.y;
    int off = blockIdx.x * 256 + t;
    float acc = 0.f;
    for (int b = 0; b < B_N; ++b)
        if (slab[b] == id) acc += feats[(size_t)b * PD + off];
    centers[(size_t)id * PD + off] = acc / fmaxf(countsf[id], 1.0f);
}

// ---------------- kernel 3: d_ap[i] (f32) and sq[i,p] (bf16-consistent) -----
__global__ void k_dap_sq(const float* __restrict__ feats, const float* __restrict__ centers,
                         const int* __restrict__ labels,
                         float* __restrict__ d_ap, float* __restrict__ sq) {
    __shared__ float2 red[256];
    int i = blockIdx.x, t = threadIdx.x;
    int lab = labels[i];
    const float* frow = feats   + (size_t)i   * PD;
    const float* crow = centers + (size_t)lab * PD;
    float dap_acc = 0.f;
    for (int p = 0; p < P_N; ++p) {
        float s2 = 0.f, sf = 0.f;
        for (int d = t; d < D_N; d += 256) {
            float fv = frow[p * D_N + d];
            float cv = crow[p * D_N + d];
            float df = fv - cv;
            s2 += df * df;
            float fbv = bf2f(f2bf(fv));        // match the GEMM's bf16 rounding
            sf += fbv * fbv;
        }
        red[t] = make_float2(s2, sf);
        __syncthreads();
        for (int s = 128; s > 0; s >>= 1) {
            if (t < s) { red[t].x += red[t + s].x; red[t].y += red[t + s].y; }
            __syncthreads();
        }
        if (t == 0) {
            dap_acc += sqrtf(fmaxf(red[0].x, EPS_F));
            sq[i * P_N + p] = red[0].y;
        }
        __syncthreads();
    }
    if (t == 0) d_ap[i] = dap_acc;
}

// ---------------- kernel 4: async double-buffered bf16 WMMA GEMM + epilogue
__global__ void __launch_bounds__(256, 1) k_pairloss(
        const unsigned short* __restrict__ featsbf, const int* __restrict__ labels,
        const float* __restrict__ d_ap, const float* __restrict__ sq,
        float* __restrict__ partials) {
    __shared__ __align__(16) unsigned short As[2][TM][LDS_STRIDE];
    __shared__ __align__(16) unsigned short Bs[2][TN][LDS_STRIDE];
    __shared__ float sqI[TM], sqJ[TN], dapI[TM], dapJ[TN];
    __shared__ int   labI[TM], labJ[TN];
    __shared__ float red[256];

    const int t    = threadIdx.x;
    const int lane = t & 31;
    const int w    = t >> 5;                   // 8 waves: 2 (M) x 4 (N)
    const int wm   = w >> 2;                   // 0..1
    const int wn   = w & 3;                    // 0..3
    const int h    = lane >> 4;
    const int ln   = lane & 15;

    const int I0 = blockIdx.y * TM;
    const int J0 = blockIdx.x * TN;

    if (t < TM) { dapI[t] = d_ap[I0 + t]; labI[t] = labels[I0 + t]; }
    {
        int q = t - 64;
        if (q >= 0 && q < TN) { dapJ[q] = d_ap[J0 + q]; labJ[q] = labels[J0 + q]; }
    }

    // Fixed per-thread async routes. Flattened tile index kt: the global source
    // of every chunk is (base + kt*TK*2 bytes) because p*D_N + k0 == kt*TK.
    unsigned voffA[2], ldsA[2], voffB[4], ldsB[4];
#pragma unroll
    for (int c0 = 0; c0 < 2; ++c0) {           // A: 512 x 16B chunks
        int c = t + c0 * 256, row = c >> 3, kc = (c & 7) * 8;
        voffA[c0] = (unsigned)(((I0 + row) * PD + kc) * 2);
        ldsA[c0]  = lds_off_of(&As[0][row][kc]);
    }
#pragma unroll
    for (int c0 = 0; c0 < 4; ++c0) {           // B: 1024 x 16B chunks
        int c = t + c0 * 256, row = c >> 3, kc = (c & 7) * 8;
        voffB[c0] = (unsigned)(((J0 + row) * PD + kc) * 2);
        ldsB[c0]  = lds_off_of(&Bs[0][row][kc]);
    }
    unsigned long long gbase = (unsigned long long)(size_t)featsbf;

    // issue one TK-wide tile pair into buffer `buf`; advances scalar base
    auto issue = [&](int buf) {
        const unsigned bA = (unsigned)buf * A_BUF_BYTES;
        const unsigned bB = (unsigned)buf * B_BUF_BYTES;
#pragma unroll
        for (int c0 = 0; c0 < 2; ++c0) async_b128_s(ldsA[c0] + bA, voffA[c0], gbase);
#pragma unroll
        for (int c0 = 0; c0 < 4; ++c0) async_b128_s(ldsB[c0] + bB, voffB[c0], gbase);
        gbase += TK * 2;                       // next k tile (rolls across p too)
    };

    float distAcc[2][2][8];
#pragma unroll
    for (int mt = 0; mt < 2; ++mt)
#pragma unroll
        for (int nt = 0; nt < 2; ++nt)
#pragma unroll
            for (int r = 0; r < 8; ++r) distAcc[mt][nt][r] = 0.f;

    const int kt_total = P_N * NKT;            // 256 tiles, pipelined across p
    issue(0);                                  // prologue

    for (int p = 0; p < P_N; ++p) {
        __syncthreads();                       // prev epilogue done before sq overwrite
        if (t < TM) sqI[t] = sq[(I0 + t) * P_N + p];
        {
            int q = t - 64;
            if (q >= 0 && q < TN) sqJ[q] = sq[(J0 + q) * P_N + p];
        }

        v8f acc[2][2];
#pragma unroll
        for (int mt = 0; mt < 2; ++mt)
#pragma unroll
            for (int nt = 0; nt < 2; ++nt) acc[mt][nt] = (v8f){};

        for (int kti = 0; kti < NKT; ++kti) {
            const int kt = p * NKT + kti;
            wait_async0();                     // my transfers for buf (kt&1) done
            __syncthreads();                   // everyone's transfers visible
            if (kt + 1 < kt_total)             // prefetch next tile into other buf
                issue((kt + 1) & 1);
            const int buf = kt & 1;
#pragma unroll
            for (int kk = 0; kk < TK; kk += 32) {
                union { v16bf v; uint4 u[2]; } fa[2], fb[2];
#pragma unroll
                for (int mt = 0; mt < 2; ++mt) {
                    fa[mt].u[0] = *(const uint4*)&As[buf][wm * 32 + mt * 16 + ln][kk + h * 8];
                    fa[mt].u[1] = *(const uint4*)&As[buf][wm * 32 + mt * 16 + ln][kk + 16 + h * 8];
                }
#pragma unroll
                for (int nt = 0; nt < 2; ++nt) {
                    fb[nt].u[0] = *(const uint4*)&Bs[buf][wn * 32 + nt * 16 + ln][kk + h * 8];
                    fb[nt].u[1] = *(const uint4*)&Bs[buf][wn * 32 + nt * 16 + ln][kk + 16 + h * 8];
                }
#pragma unroll
                for (int mt = 0; mt < 2; ++mt)
#pragma unroll
                    for (int nt = 0; nt < 2; ++nt)
                        acc[mt][nt] = __builtin_amdgcn_wmma_f32_16x16x32_bf16(
                            false, fa[mt].v, false, fb[nt].v, (short)0,
                            acc[mt][nt], false, false);
            }
        }
        // epilogue for this p: dist_aa += sqrt(clip(sq_i + sq_j - 2*inner, eps))
#pragma unroll
        for (int mt = 0; mt < 2; ++mt)
#pragma unroll
            for (int nt = 0; nt < 2; ++nt)
#pragma unroll
                for (int r = 0; r < 8; ++r) {
                    int il = wm * 32 + mt * 16 + h * 8 + r;   // C layout: M = r + h*8
                    int jl = wn * 32 + nt * 16 + ln;
                    float d2 = sqI[il] + sqJ[jl] - 2.0f * acc[mt][nt][r];
                    distAcc[mt][nt][r] += sqrtf(fmaxf(d2, EPS_F));
                }
    }

    // masked margin: relu(d_ap[i] + d_ap[j] - dist_aa + m) over different labels
    float lsum = 0.f;
#pragma unroll
    for (int mt = 0; mt < 2; ++mt)
#pragma unroll
        for (int nt = 0; nt < 2; ++nt)
#pragma unroll
            for (int r = 0; r < 8; ++r) {
                int il = wm * 32 + mt * 16 + h * 8 + r;
                int jl = wn * 32 + nt * 16 + ln;
                float v = dapI[il] + dapJ[jl] - distAcc[mt][nt][r] + MARGIN_F;
                if (labI[il] != labJ[jl] && v > 0.f) lsum += v;
            }
    __syncthreads();
    red[t] = lsum;
    __syncthreads();
    for (int s = 128; s > 0; s >>= 1) { if (t < s) red[t] += red[t + s]; __syncthreads(); }
    if (t == 0) partials[blockIdx.y * 8 + blockIdx.x] = red[0];
}

// ---------------- kernel 5: final reduction + denominator -------------------
__global__ void k_final(const float* __restrict__ partials, const float* __restrict__ countsf,
                        float* __restrict__ out) {
    __shared__ float red[128];
    int t = threadIdx.x;                       // 128 threads
    red[t] = partials[t];
    __syncthreads();
    for (int s = 64; s > 0; s >>= 1) { if (t < s) red[t] += red[t + s]; __syncthreads(); }
    if (t == 0) {
        float denom = (float)B_N * (float)B_N; // B^2 - sum(cnt^2), exact in f32
        for (int c = 0; c < NUM_IDS; ++c) denom -= countsf[c] * countsf[c];
        out[0] = red[0] / denom;
    }
}

extern "C" void kernel_launch(void* const* d_in, const int* in_sizes, int n_in,
                              void* d_out, int out_size, void* d_ws, size_t ws_size,
                              hipStream_t stream) {
    (void)in_sizes; (void)n_in; (void)out_size; (void)ws_size;
    const float* feats  = (const float*)d_in[0];
    const int*   labels = (const int*)d_in[1];
    // d_in[2] (subs) is unused by the reference.

    // workspace layout
    unsigned short* featsbf = (unsigned short*)d_ws;                     // 32 MB
    float* centers  = (float*)((char*)d_ws + (size_t)B_N * PD * 2);      // 4 MB
    float* countsf  = centers + (size_t)NUM_IDS * PD;                    // 64
    float* d_ap     = countsf + NUM_IDS;                                 // 1024
    float* sq       = d_ap + B_N;                                        // 8192
    float* partials = sq + (size_t)B_N * P_N;                            // 128

    k_tobf16  <<<(B_N * PD) / (256 * 4), 256, 0, stream>>>(feats, featsbf);
    k_counts  <<<1, 64, 0, stream>>>(labels, countsf);
    k_centers <<<dim3(64, 64), 256, 0, stream>>>(feats, labels, countsf, centers);
    k_dap_sq  <<<B_N, 256, 0, stream>>>(feats, centers, labels, d_ap, sq);
    k_pairloss<<<dim3(1024 / TN, 1024 / TM), 256, 0, stream>>>(featsbf, labels, d_ap, sq, partials);
    k_final   <<<1, 128, 0, stream>>>(partials, countsf, (float*)d_out);
}
